// SimulatorModel_77532749628021
// MI455X (gfx1250) — matrix operand order repaired and checked by reading, not compiled, not run on hardware
//
#include <hip/hip_runtime.h>
#include <math.h>

// ---------------------------------------------------------------------------
// GNN simulator for MI455X (gfx1250, wave32, WMMA + TDM).
// Edge/Node/Decoder MLPs run on v_wmma_f32_16x16x32_f16 (f16 in, f32 accum).
// 128x128 weights staged to LDS via tensor_load_to_lds (TENSORcnt).
// First/last layer weight fragments + biases are register-resident.
// Each wave processes two 16-row tiles so every LDS B-fragment feeds 2 WMMAs.
// ---------------------------------------------------------------------------

typedef _Float16 half_t;
typedef half_t v8h  __attribute__((ext_vector_type(8)));
typedef half_t v16h __attribute__((ext_vector_type(16)));
typedef float  v8f  __attribute__((ext_vector_type(8)));
typedef unsigned int u32x4 __attribute__((ext_vector_type(4)));
typedef int i32x8 __attribute__((ext_vector_type(8)));
typedef int i32x4 __attribute__((ext_vector_type(4)));

#define HIDN 128

static __device__ __forceinline__ v8f wmma32(v16h a, v16h b, v8f c) {
  return __builtin_amdgcn_wmma_f32_16x16x32_f16(false, a, false, b, (short)0, c,
                                                false, false);
}

// A fragment (16x32 f16), ISA layout:
//   lanes 0-15: M=lane,    halves 0-7 = K 0..7,  halves 8-15 = K 16..23
//   lanes16-31: M=lane-16, halves 0-7 = K 8..15, halves 8-15 = K 24..31
static __device__ __forceinline__ v16h load_a_frag(const half_t* tile,
                                                   int rowStride, int kofs) {
  int lane = (int)(threadIdx.x & 31u);
  int m  = lane & 15;
  int kb = (lane & 16) ? 8 : 0;
  const half_t* p = tile + m * rowStride + kofs + kb;
  v8h lo = *(const v8h*)(p);
  v8h hi = *(const v8h*)(p + 16);
  return __builtin_shufflevector(lo, hi, 0,1,2,3,4,5,6,7,8,9,10,11,12,13,14,15);
}

// B fragment (32x16 f16) from transposed weights wT[N][K]:
//   lanes 0-15: N=lane,    K = kofs+0..7, kofs+16..23; lanes16-31: K=kofs+8..15, kofs+24..31
static __device__ __forceinline__ v16h load_b_frag(const half_t* wT, int kStride,
                                                   int nofs, int kofs) {
  int lane = (int)(threadIdx.x & 31u);
  int n  = lane & 15;
  int kb = (lane & 16) ? 8 : 0;
  const half_t* p = wT + (size_t)(nofs + n) * kStride + kofs + kb;
  v8h lo = *(const v8h*)(p);
  v8h hi = *(const v8h*)(p + 16);
  return __builtin_shufflevector(lo, hi, 0,1,2,3,4,5,6,7,8,9,10,11,12,13,14,15);
}

#define LDS_FENCE() asm volatile("s_wait_dscnt 0" ::: "memory")

// ---- Tensor Data Mover: 1-D copy global -> LDS (nbytes multiple of 8, <=512KB)
static __device__ __forceinline__ void tdm_load_1d(void* lds_dst,
                                                   const void* gsrc,
                                                   unsigned nbytes) {
#if __has_builtin(__builtin_amdgcn_tensor_load_to_lds)
  unsigned n8 = nbytes >> 3;  // 8-byte elements (tile_dim0, 16-bit field)
  unsigned lds_off = (unsigned)(size_t)lds_dst;  // flat LDS addr[31:0] = offset
  unsigned long long ga = (unsigned long long)(size_t)gsrc;
  u32x4 g0;
  g0.x = 1u;                                   // count=1, user descriptor
  g0.y = lds_off;                              // lds_addr
  g0.z = (unsigned)ga;                         // global_addr[31:0]
  g0.w = (unsigned)((ga >> 32) & 0x01FFFFFFu) | (2u << 30);  // addr hi | type=2
  i32x8 g1;
  g1[0] = (int)(3u << 16);                     // data_size = 8B, no flags
  g1[1] = (int)((n8 & 0xFFFFu) << 16);         // tensor_dim0[15:0]
  g1[2] = (int)(((n8 >> 16) & 0xFFFFu) | (1u << 16));  // dim0 hi | tensor_dim1=1
  g1[3] = (int)((n8 & 0xFFFFu) << 16);         // tile_dim0
  g1[4] = 1;                                   // tile_dim1 = 1
  g1[5] = (int)n8;                             // tensor_dim0_stride[31:0]
  g1[6] = 0;
  g1[7] = 0;
  i32x4 zz4 = {0, 0, 0, 0};
  i32x8 zz8 = {0, 0, 0, 0, 0, 0, 0, 0};
  __builtin_amdgcn_tensor_load_to_lds(g0, g1, zz4, zz4, zz8, 0);
#else
  const uint4* s = (const uint4*)gsrc;
  uint4* d = (uint4*)lds_dst;
  int n16b = (int)(nbytes >> 4);
  for (int i = (int)(threadIdx.x & 31u); i < n16b; i += 32) d[i] = s[i];
#endif
}

static __device__ __forceinline__ void tdm_wait() {
#if __has_builtin(__builtin_amdgcn_s_wait_tensorcnt)
  __builtin_amdgcn_s_wait_tensorcnt(0);
#else
  asm volatile("s_wait_tensorcnt 0" ::: "memory");
#endif
  asm volatile("" ::: "memory");
}

// ------------------------- small utility kernels ---------------------------

__global__ void zero_f32(float* __restrict__ p, int n) {
  int i = blockIdx.x * blockDim.x + threadIdx.x;
  if (i < n) p[i] = 0.0f;
}

__global__ void build_x(const float* __restrict__ Xc, const float* __restrict__ yp,
                        float* __restrict__ x, int N) {
  int i = blockIdx.x * blockDim.x + threadIdx.x;
  if (i >= N) return;
  x[i * 5 + 0] = Xc[i * 6 + 0];
  x[i * 5 + 1] = Xc[i * 6 + 1];
  x[i * 5 + 2] = Xc[i * 6 + 2];
  x[i * 5 + 3] = Xc[i * 6 + 4];
  x[i * 5 + 4] = yp[i];
}

__global__ void init_edges(const float* __restrict__ x, const int* __restrict__ edge,
                           float* __restrict__ ea, float* __restrict__ deg, int E) {
  int e = blockIdx.x * blockDim.x + threadIdx.x;
  if (e >= E) return;
  int s = edge[e], d = edge[E + e];
  float diff = x[d * 5 + 4] - x[s * 5 + 4];
  ea[e * 3 + 0] = diff * (x[d * 5 + 0] - x[s * 5 + 0]);
  ea[e * 3 + 1] = diff * (x[d * 5 + 1] - x[s * 5 + 1]);
  ea[e * 3 + 2] = diff * (x[d * 5 + 2] - x[s * 5 + 2]);
  atomicAdd(&deg[d], 1.0f);
}

__global__ void finish_deg(const float* __restrict__ deg, float* __restrict__ inv,
                           int N) {
  int i = blockIdx.x * blockDim.x + threadIdx.x;
  if (i >= N) return;
  float dg = deg[i];
  inv[i] = (dg > 0.0f) ? (1.0f / dg) : 0.0f;
}

__global__ void aggregate(const float* __restrict__ ea, const int* __restrict__ edge,
                          float* __restrict__ aggr, int E) {
  int e = blockIdx.x * blockDim.x + threadIdx.x;
  if (e >= E) return;
  int d = edge[E + e];
  atomicAdd(&aggr[d * 3 + 0], ea[e * 3 + 0]);
  atomicAdd(&aggr[d * 3 + 1], ea[e * 3 + 1]);
  atomicAdd(&aggr[d * 3 + 2], ea[e * 3 + 2]);
}

// Pack w[K][N] (f32, row-major) -> wT[Npad][Kpad] (f16), zero padded.
__global__ void pack_wT(const float* __restrict__ w, half_t* __restrict__ out,
                        int K, int N, int Kpad, int Npad) {
  int idx = blockIdx.x * blockDim.x + threadIdx.x;
  if (idx >= Kpad * Npad) return;
  int k = idx % Kpad, n = idx / Kpad;
  float v = (k < K && n < N) ? w[k * N + n] : 0.0f;
  out[idx] = (half_t)v;
}

// ------------------------------ Edge MLP -----------------------------------
// net_in[E,9] -> 128 (relu) -> 128 (relu) -> 3 ; edge_attr += out.
// One wave per block, two 16-row tiles per iteration (32 edges).

__global__ __launch_bounds__(32) void edge_mlp(
    const float* __restrict__ x, float* __restrict__ edge_attr,
    const int* __restrict__ edge,
    const half_t* __restrict__ w1T, const half_t* __restrict__ w2T,
    const half_t* __restrict__ w3T,
    const float* __restrict__ b1, const float* __restrict__ b2,
    const float* __restrict__ b3, int E, int nPairs) {
  __shared__ __attribute__((aligned(16))) half_t sW2[HIDN * HIDN];   // 32 KB
  __shared__ __attribute__((aligned(16))) half_t actIn[32 * 32];     // 2 KB
  __shared__ __attribute__((aligned(16))) half_t act1[2 * 16 * HIDN];// 8 KB
  __shared__ __attribute__((aligned(16))) half_t act2[2 * 16 * HIDN];// 8 KB

  tdm_load_1d(sW2, w2T, HIDN * HIDN * (unsigned)sizeof(half_t));

  int lane = (int)(threadIdx.x & 31u);
  int n16 = lane & 15;
  int mofs = (lane & 16) ? 8 : 0;

  // register-resident first/last layer fragments + per-lane biases
  v16h w1f[8];
#pragma unroll
  for (int nt = 0; nt < 8; ++nt) w1f[nt] = load_b_frag(w1T, 32, nt * 16, 0);
  v16h w3f[4];
#pragma unroll
  for (int kc = 0; kc < 4; ++kc) w3f[kc] = load_b_frag(w3T, HIDN, 0, kc * 32);
  float bias1[8], bias2[8];
#pragma unroll
  for (int nt = 0; nt < 8; ++nt) {
    bias1[nt] = b1[nt * 16 + n16];
    bias2[nt] = b2[nt * 16 + n16];
  }
  float bias3 = (n16 < 3) ? b3[n16] : 0.0f;

  tdm_wait();  // sW2 resident before layer-2 use

  for (int pair = (int)blockIdx.x; pair < nPairs; pair += (int)gridDim.x) {
    int base = pair * 32;
    // ---- features: each lane builds one row (rows 0..31 = two tiles)
    {
      int e = base + lane;
      int ec = (e < E) ? e : (E - 1);
      int s = edge[ec], d = edge[E + ec];
      float xs4 = x[s * 5 + 4], xd4 = x[d * 5 + 4];
      float d0 = x[d * 5 + 0] - x[s * 5 + 0];
      float d1 = x[d * 5 + 1] - x[s * 5 + 1];
      float d2 = x[d * 5 + 2] - x[s * 5 + 2];
      float nr = sqrtf(d0 * d0 + d1 * d1 + d2 * d2);
      half_t* r = actIn + lane * 32;
      r[0] = (half_t)d0; r[1] = (half_t)d1; r[2] = (half_t)d2; r[3] = (half_t)nr;
      r[4] = (half_t)edge_attr[ec * 3 + 0];
      r[5] = (half_t)edge_attr[ec * 3 + 1];
      r[6] = (half_t)edge_attr[ec * 3 + 2];
      r[7] = (half_t)xs4; r[8] = (half_t)xd4;
#pragma unroll
      for (int k = 9; k < 32; ++k) r[k] = (half_t)0.0f;
    }
    LDS_FENCE();

    // ---- layer 1 (register B frags)
    v16h aI0 = load_a_frag(actIn, 32, 0);
    v16h aI1 = load_a_frag(actIn + 16 * 32, 32, 0);
#pragma unroll
    for (int nt = 0; nt < 8; ++nt) {
      v8f c0 = {}, c1 = {};
      c0 = wmma32(aI0, w1f[nt], c0);
      c1 = wmma32(aI1, w1f[nt], c1);
      int n = nt * 16 + n16;
#pragma unroll
      for (int r2 = 0; r2 < 8; ++r2) {
        float v0 = c0[r2] + bias1[nt];
        float v1 = c1[r2] + bias1[nt];
        act1[(r2 + mofs) * HIDN + n] = (half_t)(v0 > 0.0f ? v0 : 0.0f);
        act1[16 * HIDN + (r2 + mofs) * HIDN + n] = (half_t)(v1 > 0.0f ? v1 : 0.0f);
      }
    }
    LDS_FENCE();

    // ---- layer 2 (B frags from LDS, each feeds 2 WMMAs)
    v16h A0[4], A1[4];
#pragma unroll
    for (int kc = 0; kc < 4; ++kc) {
      A0[kc] = load_a_frag(act1, HIDN, kc * 32);
      A1[kc] = load_a_frag(act1 + 16 * HIDN, HIDN, kc * 32);
    }
#pragma unroll
    for (int nt = 0; nt < 8; ++nt) {
      v8f c0 = {}, c1 = {};
#pragma unroll
      for (int kc = 0; kc < 4; ++kc) {
        v16h b = load_b_frag(sW2, HIDN, nt * 16, kc * 32);
        c0 = wmma32(A0[kc], b, c0);
        c1 = wmma32(A1[kc], b, c1);
      }
      int n = nt * 16 + n16;
#pragma unroll
      for (int r2 = 0; r2 < 8; ++r2) {
        float v0 = c0[r2] + bias2[nt];
        float v1 = c1[r2] + bias2[nt];
        act2[(r2 + mofs) * HIDN + n] = (half_t)(v0 > 0.0f ? v0 : 0.0f);
        act2[16 * HIDN + (r2 + mofs) * HIDN + n] = (half_t)(v1 > 0.0f ? v1 : 0.0f);
      }
    }
    LDS_FENCE();

    // ---- layer 3 (register B frags), residual into edge_attr
    v8f c0 = {}, c1 = {};
#pragma unroll
    for (int kc = 0; kc < 4; ++kc) {
      c0 = wmma32(load_a_frag(act2, HIDN, kc * 32), w3f[kc], c0);
      c1 = wmma32(load_a_frag(act2 + 16 * HIDN, HIDN, kc * 32), w3f[kc], c1);
    }
    if (n16 < 3) {
#pragma unroll
      for (int r2 = 0; r2 < 8; ++r2) {
        int e0 = base + r2 + mofs;
        int e1 = base + 16 + r2 + mofs;
        if (e0 < E) edge_attr[e0 * 3 + n16] += c0[r2] + bias3;
        if (e1 < E) edge_attr[e1 * 3 + n16] += c1[r2] + bias3;
      }
    }
  }
}

// ------------------------------ Node MLP -----------------------------------

__global__ __launch_bounds__(32) void node_mlp(
    float* __restrict__ x, const float* __restrict__ aggr,
    const float* __restrict__ inv_deg,
    const half_t* __restrict__ w1T, const half_t* __restrict__ w2T,
    const half_t* __restrict__ w3T,
    const float* __restrict__ b1, const float* __restrict__ b2,
    const float* __restrict__ b3, int N, int nPairs) {
  __shared__ __attribute__((aligned(16))) half_t sW2[HIDN * HIDN];
  __shared__ __attribute__((aligned(16))) half_t actIn[32 * 32];
  __shared__ __attribute__((aligned(16))) half_t act1[2 * 16 * HIDN];
  __shared__ __attribute__((aligned(16))) half_t act2[2 * 16 * HIDN];

  tdm_load_1d(sW2, w2T, HIDN * HIDN * (unsigned)sizeof(half_t));

  int lane = (int)(threadIdx.x & 31u);
  int n16 = lane & 15;
  int mofs = (lane & 16) ? 8 : 0;

  v16h w1f[8];
#pragma unroll
  for (int nt = 0; nt < 8; ++nt) w1f[nt] = load_b_frag(w1T, 32, nt * 16, 0);
  v16h w3f[4];
#pragma unroll
  for (int kc = 0; kc < 4; ++kc) w3f[kc] = load_b_frag(w3T, HIDN, 0, kc * 32);
  float bias1[8], bias2[8];
#pragma unroll
  for (int nt = 0; nt < 8; ++nt) {
    bias1[nt] = b1[nt * 16 + n16];
    bias2[nt] = b2[nt * 16 + n16];
  }
  float bias3 = b3[0];

  tdm_wait();

  for (int pair = (int)blockIdx.x; pair < nPairs; pair += (int)gridDim.x) {
    int base = pair * 32;
    {
      int i = base + lane;
      int ic = (i < N) ? i : (N - 1);
      float inv = inv_deg[ic];
      half_t* r = actIn + lane * 32;
      r[0] = (half_t)x[ic * 5 + 3];
      r[1] = (half_t)x[ic * 5 + 4];
      r[2] = (half_t)(aggr[ic * 3 + 0] * inv);
      r[3] = (half_t)(aggr[ic * 3 + 1] * inv);
      r[4] = (half_t)(aggr[ic * 3 + 2] * inv);
#pragma unroll
      for (int k = 5; k < 32; ++k) r[k] = (half_t)0.0f;
    }
    LDS_FENCE();

    v16h aI0 = load_a_frag(actIn, 32, 0);
    v16h aI1 = load_a_frag(actIn + 16 * 32, 32, 0);
#pragma unroll
    for (int nt = 0; nt < 8; ++nt) {
      v8f c0 = {}, c1 = {};
      c0 = wmma32(aI0, w1f[nt], c0);
      c1 = wmma32(aI1, w1f[nt], c1);
      int n = nt * 16 + n16;
#pragma unroll
      for (int r2 = 0; r2 < 8; ++r2) {
        float v0 = c0[r2] + bias1[nt];
        float v1 = c1[r2] + bias1[nt];
        act1[(r2 + mofs) * HIDN + n] = (half_t)(v0 > 0.0f ? v0 : 0.0f);
        act1[16 * HIDN + (r2 + mofs) * HIDN + n] = (half_t)(v1 > 0.0f ? v1 : 0.0f);
      }
    }
    LDS_FENCE();

    v16h A0[4], A1[4];
#pragma unroll
    for (int kc = 0; kc < 4; ++kc) {
      A0[kc] = load_a_frag(act1, HIDN, kc * 32);
      A1[kc] = load_a_frag(act1 + 16 * HIDN, HIDN, kc * 32);
    }
#pragma unroll
    for (int nt = 0; nt < 8; ++nt) {
      v8f c0 = {}, c1 = {};
#pragma unroll
      for (int kc = 0; kc < 4; ++kc) {
        v16h b = load_b_frag(sW2, HIDN, nt * 16, kc * 32);
        c0 = wmma32(A0[kc], b, c0);
        c1 = wmma32(A1[kc], b, c1);
      }
      int n = nt * 16 + n16;
#pragma unroll
      for (int r2 = 0; r2 < 8; ++r2) {
        float v0 = c0[r2] + bias2[nt];
        float v1 = c1[r2] + bias2[nt];
        act2[(r2 + mofs) * HIDN + n] = (half_t)(v0 > 0.0f ? v0 : 0.0f);
        act2[16 * HIDN + (r2 + mofs) * HIDN + n] = (half_t)(v1 > 0.0f ? v1 : 0.0f);
      }
    }
    LDS_FENCE();

    v8f c0 = {}, c1 = {};
#pragma unroll
    for (int kc = 0; kc < 4; ++kc) {
      c0 = wmma32(load_a_frag(act2, HIDN, kc * 32), w3f[kc], c0);
      c1 = wmma32(load_a_frag(act2 + 16 * HIDN, HIDN, kc * 32), w3f[kc], c1);
    }
    if (n16 == 0) {
#pragma unroll
      for (int r2 = 0; r2 < 8; ++r2) {
        int i0 = base + r2 + mofs;
        int i1 = base + 16 + r2 + mofs;
        if (i0 < N) x[i0 * 5 + 4] += c0[r2] + bias3;
        if (i1 < N) x[i1 * 5 + 4] += c1[r2] + bias3;
      }
    }
  }
}

// ------------------------------ Decoder ------------------------------------
// x[N,5] -> 128 -> 128 -> 128 -> 1 ; out = y_prev + res.
// dw2 staged via TDM; dw3 fragments streamed from global (L2-resident).

__global__ __launch_bounds__(32) void decoder_mlp(
    const float* __restrict__ x, const float* __restrict__ y_prev,
    const half_t* __restrict__ w1T, const half_t* __restrict__ w2T,
    const half_t* __restrict__ w3T, const half_t* __restrict__ w4T,
    const float* __restrict__ b1, const float* __restrict__ b2,
    const float* __restrict__ b3, const float* __restrict__ b4,
    float* __restrict__ out, int N, int nPairs) {
  __shared__ __attribute__((aligned(16))) half_t sW2[HIDN * HIDN];
  __shared__ __attribute__((aligned(16))) half_t actIn[32 * 32];
  __shared__ __attribute__((aligned(16))) half_t act1[2 * 16 * HIDN];
  __shared__ __attribute__((aligned(16))) half_t act2[2 * 16 * HIDN];

  tdm_load_1d(sW2, w2T, HIDN * HIDN * (unsigned)sizeof(half_t));

  int lane = (int)(threadIdx.x & 31u);
  int n16 = lane & 15;
  int mofs = (lane & 16) ? 8 : 0;

  v16h w1f[8];
#pragma unroll
  for (int nt = 0; nt < 8; ++nt) w1f[nt] = load_b_frag(w1T, 32, nt * 16, 0);
  v16h w4f[4];
#pragma unroll
  for (int kc = 0; kc < 4; ++kc) w4f[kc] = load_b_frag(w4T, HIDN, 0, kc * 32);
  float bias1[8], bias2[8], bias3[8];
#pragma unroll
  for (int nt = 0; nt < 8; ++nt) {
    bias1[nt] = b1[nt * 16 + n16];
    bias2[nt] = b2[nt * 16 + n16];
    bias3[nt] = b3[nt * 16 + n16];
  }
  float bias4 = b4[0];

  tdm_wait();

  for (int pair = (int)blockIdx.x; pair < nPairs; pair += (int)gridDim.x) {
    int base = pair * 32;
    {
      int i = base + lane;
      int ic = (i < N) ? i : (N - 1);
      half_t* r = actIn + lane * 32;
#pragma unroll
      for (int k = 0; k < 5; ++k) r[k] = (half_t)x[ic * 5 + k];
#pragma unroll
      for (int k = 5; k < 32; ++k) r[k] = (half_t)0.0f;
    }
    LDS_FENCE();

    // L1
    v16h aI0 = load_a_frag(actIn, 32, 0);
    v16h aI1 = load_a_frag(actIn + 16 * 32, 32, 0);
#pragma unroll
    for (int nt = 0; nt < 8; ++nt) {
      v8f c0 = {}, c1 = {};
      c0 = wmma32(aI0, w1f[nt], c0);
      c1 = wmma32(aI1, w1f[nt], c1);
      int n = nt * 16 + n16;
#pragma unroll
      for (int r2 = 0; r2 < 8; ++r2) {
        float v0 = c0[r2] + bias1[nt];
        float v1 = c1[r2] + bias1[nt];
        act1[(r2 + mofs) * HIDN + n] = (half_t)(v0 > 0.0f ? v0 : 0.0f);
        act1[16 * HIDN + (r2 + mofs) * HIDN + n] = (half_t)(v1 > 0.0f ? v1 : 0.0f);
      }
    }
    LDS_FENCE();

    // L2: act1 -> act2 (sW2 from LDS)
    v16h A0[4], A1[4];
#pragma unroll
    for (int kc = 0; kc < 4; ++kc) {
      A0[kc] = load_a_frag(act1, HIDN, kc * 32);
      A1[kc] = load_a_frag(act1 + 16 * HIDN, HIDN, kc * 32);
    }
#pragma unroll
    for (int nt = 0; nt < 8; ++nt) {
      v8f c0 = {}, c1 = {};
#pragma unroll
      for (int kc = 0; kc < 4; ++kc) {
        v16h b = load_b_frag(sW2, HIDN, nt * 16, kc * 32);
        c0 = wmma32(A0[kc], b, c0);
        c1 = wmma32(A1[kc], b, c1);
      }
      int n = nt * 16 + n16;
#pragma unroll
      for (int r2 = 0; r2 < 8; ++r2) {
        float v0 = c0[r2] + bias2[nt];
        float v1 = c1[r2] + bias2[nt];
        act2[(r2 + mofs) * HIDN + n] = (half_t)(v0 > 0.0f ? v0 : 0.0f);
        act2[16 * HIDN + (r2 + mofs) * HIDN + n] = (half_t)(v1 > 0.0f ? v1 : 0.0f);
      }
    }
    LDS_FENCE();

    // L3: act2 -> act1 (w3T from global / L2)
#pragma unroll
    for (int kc = 0; kc < 4; ++kc) {
      A0[kc] = load_a_frag(act2, HIDN, kc * 32);
      A1[kc] = load_a_frag(act2 + 16 * HIDN, HIDN, kc * 32);
    }
#pragma unroll
    for (int nt = 0; nt < 8; ++nt) {
      v8f c0 = {}, c1 = {};
#pragma unroll
      for (int kc = 0; kc < 4; ++kc) {
        v16h b = load_b_frag(w3T, HIDN, nt * 16, kc * 32);
        c0 = wmma32(A0[kc], b, c0);
        c1 = wmma32(A1[kc], b, c1);
      }
      int n = nt * 16 + n16;
#pragma unroll
      for (int r2 = 0; r2 < 8; ++r2) {
        float v0 = c0[r2] + bias3[nt];
        float v1 = c1[r2] + bias3[nt];
        act1[(r2 + mofs) * HIDN + n] = (half_t)(v0 > 0.0f ? v0 : 0.0f);
        act1[16 * HIDN + (r2 + mofs) * HIDN + n] = (half_t)(v1 > 0.0f ? v1 : 0.0f);
      }
    }
    LDS_FENCE();

    // L4
    v8f c0 = {}, c1 = {};
#pragma unroll
    for (int kc = 0; kc < 4; ++kc) {
      c0 = wmma32(load_a_frag(act1, HIDN, kc * 32), w4f[kc], c0);
      c1 = wmma32(load_a_frag(act1 + 16 * HIDN, HIDN, kc * 32), w4f[kc], c1);
    }
    if (n16 == 0) {
#pragma unroll
      for (int r2 = 0; r2 < 8; ++r2) {
        int i0 = base + r2 + mofs;
        int i1 = base + 16 + r2 + mofs;
        if (i0 < N) out[i0] = y_prev[i0] + c0[r2] + bias4;
        if (i1 < N) out[i1] = y_prev[i1] + c1[r2] + bias4;
      }
    }
  }
}

// ------------------------------- launcher ----------------------------------

extern "C" void kernel_launch(void* const* d_in, const int* in_sizes, int n_in,
                              void* d_out, int out_size, void* d_ws,
                              size_t ws_size, hipStream_t stream) {
  (void)n_in; (void)out_size; (void)ws_size;
  const float* X_curr = (const float*)d_in[0];
  const float* y_prev = (const float*)d_in[1];
  const int*   edge   = (const int*)d_in[2];
  const float* ew1 = (const float*)d_in[4],  *eb1 = (const float*)d_in[5];
  const float* ew2 = (const float*)d_in[6],  *eb2 = (const float*)d_in[7];
  const float* ew3 = (const float*)d_in[8],  *eb3 = (const float*)d_in[9];
  const float* nw1 = (const float*)d_in[10], *nb1 = (const float*)d_in[11];
  const float* nw2 = (const float*)d_in[12], *nb2 = (const float*)d_in[13];
  const float* nw3 = (const float*)d_in[14], *nb3 = (const float*)d_in[15];
  const float* dw1 = (const float*)d_in[16], *db1 = (const float*)d_in[17];
  const float* dw2 = (const float*)d_in[18], *db2 = (const float*)d_in[19];
  const float* dw3 = (const float*)d_in[20], *db3 = (const float*)d_in[21];
  const float* dw4 = (const float*)d_in[22], *db4 = (const float*)d_in[23];
  int N = in_sizes[0] / 6;
  int E = in_sizes[2] / 2;
  float* out = (float*)d_out;

  char* ws = (char*)d_ws;
  size_t off = 0;
  auto carve = [&](size_t bytes) -> char* {
    off = (off + 255) & ~(size_t)255;
    char* p = ws + off;
    off += bytes;
    return p;
  };
  float* x       = (float*)carve((size_t)N * 5 * 4);
  float* eattr   = (float*)carve((size_t)E * 3 * 4);
  float* aggr    = (float*)carve((size_t)N * 3 * 4);
  float* deg     = (float*)carve((size_t)N * 4);
  float* inv_deg = (float*)carve((size_t)N * 4);
  half_t* ew1T = (half_t*)carve(HIDN * 32 * 2);
  half_t* ew2T = (half_t*)carve(HIDN * HIDN * 2);
  half_t* ew3T = (half_t*)carve(16 * HIDN * 2);
  half_t* nw1T = (half_t*)carve(HIDN * 32 * 2);
  half_t* nw2T = (half_t*)carve(HIDN * HIDN * 2);
  half_t* nw3T = (half_t*)carve(16 * HIDN * 2);
  half_t* dw1T = (half_t*)carve(HIDN * 32 * 2);
  half_t* dw2T = (half_t*)carve(HIDN * HIDN * 2);
  half_t* dw3T = (half_t*)carve(HIDN * HIDN * 2);
  half_t* dw4T = (half_t*)carve(16 * HIDN * 2);

  const int TPB = 256;
#define GRID1(n) (((n) + TPB - 1) / TPB)

  pack_wT<<<GRID1(32 * HIDN), TPB, 0, stream>>>(ew1, ew1T, 9, HIDN, 32, HIDN);
  pack_wT<<<GRID1(HIDN * HIDN), TPB, 0, stream>>>(ew2, ew2T, HIDN, HIDN, HIDN, HIDN);
  pack_wT<<<GRID1(HIDN * 16), TPB, 0, stream>>>(ew3, ew3T, HIDN, 3, HIDN, 16);
  pack_wT<<<GRID1(32 * HIDN), TPB, 0, stream>>>(nw1, nw1T, 5, HIDN, 32, HIDN);
  pack_wT<<<GRID1(HIDN * HIDN), TPB, 0, stream>>>(nw2, nw2T, HIDN, HIDN, HIDN, HIDN);
  pack_wT<<<GRID1(HIDN * 16), TPB, 0, stream>>>(nw3, nw3T, HIDN, 1, HIDN, 16);
  pack_wT<<<GRID1(32 * HIDN), TPB, 0, stream>>>(dw1, dw1T, 5, HIDN, 32, HIDN);
  pack_wT<<<GRID1(HIDN * HIDN), TPB, 0, stream>>>(dw2, dw2T, HIDN, HIDN, HIDN, HIDN);
  pack_wT<<<GRID1(HIDN * HIDN), TPB, 0, stream>>>(dw3, dw3T, HIDN, HIDN, HIDN, HIDN);
  pack_wT<<<GRID1(HIDN * 16), TPB, 0, stream>>>(dw4, dw4T, HIDN, 1, HIDN, 16);

  build_x<<<GRID1(N), TPB, 0, stream>>>(X_curr, y_prev, x, N);
  zero_f32<<<GRID1(N), TPB, 0, stream>>>(deg, N);
  init_edges<<<GRID1(E), TPB, 0, stream>>>(x, edge, eattr, deg, E);
  finish_deg<<<GRID1(N), TPB, 0, stream>>>(deg, inv_deg, N);

  int ePairs = (E + 31) / 32;
  int nPairs = (N + 31) / 32;
  int eBlocks = ePairs < 4096 ? ePairs : 4096;
  int nBlocks = nPairs < 2048 ? nPairs : 2048;

  for (int it = 0; it < 3; ++it) {
    edge_mlp<<<eBlocks, 32, 0, stream>>>(x, eattr, edge, ew1T, ew2T, ew3T,
                                         eb1, eb2, eb3, E, ePairs);
    zero_f32<<<GRID1(N * 3), TPB, 0, stream>>>(aggr, N * 3);
    aggregate<<<GRID1(E), TPB, 0, stream>>>(eattr, edge, aggr, E);
    node_mlp<<<nBlocks, 32, 0, stream>>>(x, aggr, inv_deg, nw1T, nw2T, nw3T,
                                         nb1, nb2, nb3, N, nPairs);
  }

  decoder_mlp<<<nBlocks, 32, 0, stream>>>(x, y_prev, dw1T, dw2T, dw3T, dw4T,
                                          db1, db2, db3, db4, out, N, nPairs);
}
